// GNNModel_6923487281348
// MI455X (gfx1250) — matrix operand order repaired and checked by reading, not compiled, not run on hardware
//
#include <hip/hip_runtime.h>
#include <math.h>

#define NN   1024
#define EMB  128
#define FC   512
#define ETOT 523776   // NN*(NN-1)/2
#define MT   64       // edges per block (ETOT = 8184 * 64 exactly)
#define ES   136      // e-tile LDS stride in halves (128 + 8 pad, 16B-aligned rows)
#define OS   520      // o1 LDS stride in halves   (512 + 8 pad, 16B-aligned rows)

typedef __attribute__((ext_vector_type(16))) _Float16     v16h;
typedef __attribute__((ext_vector_type(8)))  float        v8f;
typedef __attribute__((ext_vector_type(4)))  unsigned int v4u;
typedef __attribute__((ext_vector_type(4)))  float        v4f;

union V16 { v16h v; v4u q[2]; };

// ---------------- SAGE front-end (tiny, VALU is fine) ----------------

__global__ void k_mean_x(const float* __restrict__ x, float* __restrict__ agg0) {
  __shared__ float s2[2];
  int t = threadIdx.x;
  if (t < 2) s2[t] = 0.f;
  __syncthreads();
  int col = t & 1;
  float p = 0.f;
  for (int r = t >> 1; r < NN; r += 32) p += x[r * 2 + col];
  atomicAdd(&s2[col], p);
  __syncthreads();
  if (t < 2) agg0[t] = s2[t] * (1.f / NN);
}

__global__ void k_h0(const float* __restrict__ x, const float* __restrict__ w_rel0,
                     const float* __restrict__ b_rel0, const float* __restrict__ w_root0,
                     const float* __restrict__ agg0, float* __restrict__ h0) {
  int g = blockIdx.x * blockDim.x + threadIdx.x;
  if (g >= NN * EMB) return;
  int i = g >> 7, j = g & 127;
  float v = agg0[0] * w_rel0[j * 2] + agg0[1] * w_rel0[j * 2 + 1] + b_rel0[j]
          + x[i * 2] * w_root0[j * 2] + x[i * 2 + 1] * w_root0[j * 2 + 1];
  h0[g] = v > 0.f ? v : 0.f;
}

// one block of 128 threads: column means of h0, then relpart[j] = agg1 . w_rel1[j] + b
__global__ void k_relpart(const float* __restrict__ h0, const float* __restrict__ w_rel1,
                          const float* __restrict__ b_rel1, float* __restrict__ relpart) {
  __shared__ float sa[EMB];
  int j = threadIdx.x;
  float s = 0.f;
  for (int r = 0; r < NN; ++r) s += h0[r * EMB + j];
  sa[j] = s * (1.f / NN);
  __syncthreads();
  float d = b_rel1[j];
  for (int k = 0; k < EMB; ++k) d += sa[k] * w_rel1[j * EMB + k];
  relpart[j] = d;
}

__global__ void k_h1(const float* __restrict__ h0, const float* __restrict__ w_root1,
                     const float* __restrict__ relpart, float* __restrict__ h1) {
  __shared__ float row[EMB];
  int i = blockIdx.x, j = threadIdx.x;
  row[j] = h0[i * EMB + j];
  __syncthreads();
  float d = relpart[j];
  for (int k = 0; k < EMB; ++k) d += row[k] * w_root1[j * EMB + k];
  h1[i * EMB + j] = d > 0.f ? d : 0.f;
}

// fp32 -> fp16 weight conversion (done every call; deterministic, tiny)
__global__ void k_cvt(const float* __restrict__ lw0, const float* __restrict__ lw1,
                      _Float16* __restrict__ W0h, _Float16* __restrict__ W1h) {
  int g = blockIdx.x * blockDim.x + threadIdx.x;
  if (g < FC * EMB) W0h[g] = (_Float16)lw0[g];
  int g2 = g - FC * EMB;
  if (g2 >= 0 && g2 < FC * FC) W1h[g2] = (_Float16)lw1[g2];
}

// ---------------- edge MLP: the WMMA heavy hitter ----------------
// 512 threads = 16 waves. Wave w owns N-tiles {2w, 2w+1} of the 512-wide layers.
// Per k-step: one B (32x16 f16, two global_load_b128 / lane) is reused by the
// 4 M-tiles; all 4 A tiles (16x32 f16) are loaded from LDS into distinct
// registers FIRST (8 ds_load_b128 clause + single wait), then 4 back-to-back
// WMMAs — keeps A-load latency off the WMMA critical path.

__global__ __launch_bounds__(512) void k_edge_mlp(
    const float* __restrict__ h1,
    const _Float16* __restrict__ W0h,   // FC x EMB
    const _Float16* __restrict__ W1h,   // FC x FC
    const float* __restrict__ lb0,
    const float* __restrict__ lb1,
    const float* __restrict__ lw2,
    const float* __restrict__ lb2,
    float* __restrict__ out)
{
  extern __shared__ char smem[];
  _Float16* eT   = (_Float16*)(smem);            // 64*136*2 = 17408 B
  _Float16* o1s  = (_Float16*)(smem + 17408);    // 64*520*2 = 66560 B
  int*      sEi  = (int*)(smem + 83968);         // 256 B
  int*      sEj  = (int*)(smem + 84224);         // 256 B
  float*    sRed = (float*)(smem + 84480);       // 256 B

  const int tid    = threadIdx.x;
  const int lane   = tid & 31;
  const int wave   = tid >> 5;
  const int lane15 = lane & 15;
  const int laneHi = lane >> 4;
  const long eBase = (long)blockIdx.x * MT;

  // --- phase 0a: edge index inversion + reduction init ---
  if (tid < MT) {
    long L = eBase + tid;
    double a = 2.0 * NN - 1.0;
    long i = (long)((a - sqrt(a * a - 8.0 * (double)L)) * 0.5);
    if (i < 0) i = 0;
    if (i > NN - 2) i = NN - 2;
    while (i < NN - 2 && (i + 1) * (2L * NN - 1 - (i + 1)) / 2 <= L) ++i;
    while (i > 0 && i * (2L * NN - 1 - i) / 2 > L) --i;
    long j = i + 1 + (L - i * (2L * NN - 1 - i) / 2);
    sEi[tid] = (int)i;
    sEj[tid] = (int)j;
    sRed[tid] = 0.f;
  }
  __syncthreads();

  // --- phase 0b: build e = h[i]*h[j] tile (64 x 128 f16) in LDS ---
  {
    int r  = tid >> 3;
    int c0 = (tid & 7) * 16;
    const v4f* pi = (const v4f*)(h1 + (size_t)sEi[r] * EMB + c0);
    const v4f* pj = (const v4f*)(h1 + (size_t)sEj[r] * EMB + c0);
    V16 pk;
    _Float16* ph = (_Float16*)&pk;
#pragma unroll
    for (int q = 0; q < 4; ++q) {
      v4f ai = pi[q], aj = pj[q];
      ph[q * 4 + 0] = (_Float16)(ai.x * aj.x);
      ph[q * 4 + 1] = (_Float16)(ai.y * aj.y);
      ph[q * 4 + 2] = (_Float16)(ai.z * aj.z);
      ph[q * 4 + 3] = (_Float16)(ai.w * aj.w);
    }
    v4u* sp = (v4u*)(eT + (size_t)r * ES + c0);
    sp[0] = pk.q[0];
    sp[1] = pk.q[1];
  }
  __syncthreads();

  // --- phase 1: o1 = relu(e @ W0^T + b0), 64x512 into LDS ---
#pragma unroll
  for (int nn = 0; nn < 2; ++nn) {
    const int   colB = (wave * 2 + nn) * 16 + lane15;
    const float b0v  = lb0[colB];
    v8f acc[4] = {};
    const _Float16* bptr = W0h + (size_t)colB * EMB + laneHi * 8;
    for (int kk = 0; kk < 4; ++kk) {
      const int k0 = kk * 32;
      V16 B;
      const v4u* bp = (const v4u*)(bptr + k0);
      B.q[0] = bp[0];
      B.q[1] = bp[2];
      V16 A[4];
#pragma unroll
      for (int mt = 0; mt < 4; ++mt) {
        const v4u* ap = (const v4u*)(eT + (size_t)(mt * 16 + lane15) * ES + k0 + laneHi * 8);
        A[mt].q[0] = ap[0];
        A[mt].q[1] = ap[2];
      }
#pragma unroll
      for (int mt = 0; mt < 4; ++mt)
        acc[mt] = __builtin_amdgcn_wmma_f32_16x16x32_f16(
            false, A[mt].v, false, B.v, (short)0, acc[mt], false, false);
    }
#pragma unroll
    for (int mt = 0; mt < 4; ++mt)
#pragma unroll
      for (int v = 0; v < 8; ++v) {
        float val = acc[mt][v] + b0v;
        val = val > 0.f ? val : 0.f;
        o1s[(size_t)(mt * 16 + v + laneHi * 8) * OS + colB] = (_Float16)val;
      }
  }
  __syncthreads();

  // --- phase 2: o2 = relu(o1 @ W1^T + b1); fold GEMM3 (dot with lw2) ---
#pragma unroll
  for (int nn = 0; nn < 2; ++nn) {
    const int   colB = (wave * 2 + nn) * 16 + lane15;
    const float b1v  = lb1[colB];
    const float w2v  = lw2[colB];
    v8f acc[4] = {};
    const _Float16* bptr = W1h + (size_t)colB * FC + laneHi * 8;
    for (int kk = 0; kk < 16; ++kk) {
      const int k0 = kk * 32;
      V16 B;
      const v4u* bp = (const v4u*)(bptr + k0);
      B.q[0] = bp[0];
      B.q[1] = bp[2];
      V16 A[4];
#pragma unroll
      for (int mt = 0; mt < 4; ++mt) {
        const v4u* ap = (const v4u*)(o1s + (size_t)(mt * 16 + lane15) * OS + k0 + laneHi * 8);
        A[mt].q[0] = ap[0];
        A[mt].q[1] = ap[2];
      }
#pragma unroll
      for (int mt = 0; mt < 4; ++mt)
        acc[mt] = __builtin_amdgcn_wmma_f32_16x16x32_f16(
            false, A[mt].v, false, B.v, (short)0, acc[mt], false, false);
    }
#pragma unroll
    for (int mt = 0; mt < 4; ++mt) {
      float p[8];
#pragma unroll
      for (int v = 0; v < 8; ++v) {
        float val = acc[mt][v] + b1v;
        val = val > 0.f ? val : 0.f;
        p[v] = val * w2v;
      }
#pragma unroll
      for (int mask = 1; mask < 16; mask <<= 1)
#pragma unroll
        for (int v = 0; v < 8; ++v)
          p[v] += __shfl_xor(p[v], mask, 16);
      if (lane15 == 0)
#pragma unroll
        for (int v = 0; v < 8; ++v)
          atomicAdd(&sRed[mt * 16 + v + laneHi * 8], p[v]);
    }
  }
  __syncthreads();

  // --- phase 3: sigmoid + store ---
  if (tid < MT) {
    float z = sRed[tid] + lb2[0];
    out[eBase + tid] = 1.f / (1.f + __expf(-z));
  }
}

// ---------------- launcher ----------------

extern "C" void kernel_launch(void* const* d_in, const int* in_sizes, int n_in,
                              void* d_out, int out_size, void* d_ws, size_t ws_size,
                              hipStream_t stream) {
  const float* x       = (const float*)d_in[0];
  const float* w_rel0  = (const float*)d_in[1];
  const float* b_rel0  = (const float*)d_in[2];
  const float* w_root0 = (const float*)d_in[3];
  const float* w_rel1  = (const float*)d_in[4];
  const float* b_rel1  = (const float*)d_in[5];
  const float* w_root1 = (const float*)d_in[6];
  const float* lw0     = (const float*)d_in[7];
  const float* lb0     = (const float*)d_in[8];
  const float* lw1     = (const float*)d_in[9];
  const float* lb1     = (const float*)d_in[10];
  const float* lw2     = (const float*)d_in[11];
  const float* lb2     = (const float*)d_in[12];
  float* out = (float*)d_out;

  char* ws = (char*)d_ws;
  float*    agg0    = (float*)(ws);                                        // 256 B
  float*    h0      = (float*)(ws + 256);                                  // 512 KB
  float*    relpart = (float*)(ws + 256 + 524288);                         // 512 B
  float*    h1      = (float*)(ws + 256 + 524288 + 512);                   // 512 KB
  _Float16* W0h     = (_Float16*)(ws + 256 + 524288 + 512 + 524288);       // 128 KB
  _Float16* W1h     = (_Float16*)(ws + 256 + 524288 + 512 + 524288 + 131072); // 512 KB

  k_mean_x<<<1, 64, 0, stream>>>(x, agg0);
  k_h0<<<(NN * EMB) / 256, 256, 0, stream>>>(x, w_rel0, b_rel0, w_root0, agg0, h0);
  k_relpart<<<1, EMB, 0, stream>>>(h0, w_rel1, b_rel1, relpart);
  k_h1<<<NN, EMB, 0, stream>>>(h0, w_root1, relpart, h1);
  k_cvt<<<(FC * EMB + FC * FC + 255) / 256, 256, 0, stream>>>(lw0, lw1, W0h, W1h);
  k_edge_mlp<<<ETOT / MT, 512, 84992, stream>>>(h1, W0h, W1h, lb0, lb1, lw2, lb2, out);
}
